// GLHFE_CSGVD_85066122265502
// MI455X (gfx1250) — matrix operand order repaired and checked by reading, not compiled, not run on hardware
//
#include <hip/hip_runtime.h>

typedef __attribute__((ext_vector_type(16))) __bf16 v16bf;
typedef __attribute__((ext_vector_type(8)))  __bf16 v8bf;
typedef __attribute__((ext_vector_type(4)))  __bf16 v4bf;
typedef __attribute__((ext_vector_type(8)))  float  v8f;

#define LDIM 128
#define TILE 16

// ---------------- WMMA fragment loaders ----------------

// A: 16x32 bf16 tile from LDS, row-major rows of 128. Per ISA 7.12.2 the 16
// elements of a lane are two contiguous 8-element K-runs:
//   a[0..7]  = K = kb*32 + grp + 0..7      (grp = 8 for lanes 16-31)
//   a[8..15] = K = kb*32 + grp + 16..23
// "lo" holds rows for K in [0,128), "hi" for K in [128,256).
__device__ __forceinline__ v16bf load_a_frag(const __bf16* lo, const __bf16* hi,
                                             int M, int kb, int laneHi) {
  int grp = laneHi ? 8 : 0;
  int K0 = kb * 32 + grp;
  int K1 = K0 + 16;
  const __bf16* p0 = ((K0 < LDIM) ? lo : hi) + M * LDIM + (K0 & (LDIM - 1));
  const __bf16* p1 = ((K1 < LDIM) ? lo : hi) + M * LDIM + (K1 & (LDIM - 1));
  v8bf l8 = *(const v8bf*)p0;   // 16B ds load
  v8bf h8 = *(const v8bf*)p1;   // 16B ds load
  return __builtin_shufflevector(l8, h8, 0, 1, 2, 3, 4, 5, 6, 7,
                                 8, 9, 10, 11, 12, 13, 14, 15);
}

// B tiles are pre-swizzled fragment-major: frag(tile)[lane][16 elems] contiguous.
// One 32-byte load per lane -> 2x global_load_b128.
__device__ __forceinline__ v16bf load_b_frag_sw(const __bf16* __restrict__ Bsw,
                                                int tileIdx, int lane) {
  return *(const v16bf*)(Bsw + ((size_t)tileIdx * 32 + lane) * 16);
}

// ---------------- Weight preparation ----------------

// Swizzled store position for B element (K-row r within 32-tile, column j):
//   lane = ((r>>3)&1)*16 + (j&15), elem = (r&7) + (r>=16 ? 8:0), ct = j>>4
__device__ __forceinline__ size_t b_swizzle_index(int tileBase, int r, int j) {
  int lane = (((r >> 3) & 1) ? 16 : 0) + (j & 15);
  int elem = (r & 7) + ((r >= 16) ? 8 : 0);
  int ct = j >> 4;
  return ((size_t)(tileBase + ct) * 32 + lane) * 16 + elem;
}

// Wk[b][k][j] = sum_o W_bases[b][k][o] * wk_w[o][j]   (bf16, swizzled)
// Wv[b][k]    = sum_o W_bases[b][k][o] * wv_w[o]      (f32)
__global__ __launch_bounds__(128) void build_wk(const float* __restrict__ Wbases,
                                                const float* __restrict__ wk_w,
                                                const float* __restrict__ wv_w,
                                                __bf16* __restrict__ WkSw,
                                                float* __restrict__ Wv) {
  int bk = blockIdx.x;            // 0..511  -> (basis, k-row)
  int j = threadIdx.x;            // 0..127
  int b = bk >> 8, k = bk & 255;
  const float* Wrow = Wbases + (size_t)bk * LDIM;
  float acc = 0.f;
  for (int o = 0; o < LDIM; ++o) acc += Wrow[o] * wk_w[o * LDIM + j];
  int kb = k >> 5, r = k & 31;
  WkSw[b_swizzle_index((b * 8 + kb) * 8, r, j)] = (__bf16)acc;
  if (j == 0) {
    float av = 0.f;
    for (int o = 0; o < LDIM; ++o) av += Wrow[o] * wv_w[o];
    Wv[bk] = av;
  }
}

// swizzled bf16 copy of wq_w; v_self[i] = sum_b coef[4][b]*(Wv[b][i]+Wv[b][i+128])
__global__ __launch_bounds__(256) void prep_misc(const float* __restrict__ wq_w,
                                                 __bf16* __restrict__ wqSw,
                                                 const float* __restrict__ Wv,
                                                 const float* __restrict__ coef,
                                                 float* __restrict__ v_self) {
  int i = blockIdx.x * 256 + threadIdx.x;
  if (i < LDIM * LDIM) {
    int o = i >> 7, j = i & (LDIM - 1);        // wq_w[o][j], K-row = o
    int kb = o >> 5, r = o & 31;
    wqSw[b_swizzle_index(kb * 8, r, j)] = (__bf16)wq_w[i];
  }
  if (i < LDIM) {
    v_self[i] = coef[4 * 2 + 0] * (Wv[i] + Wv[LDIM + i]) +
                coef[4 * 2 + 1] * (Wv[256 + i] + Wv[256 + LDIM + i]);
  }
}

// ---------------- n_q = h @ wq_w + wq_b  (WMMA) ----------------

__global__ __launch_bounds__(128) void node_q_kernel(const float* __restrict__ h,
                                                     const __bf16* __restrict__ wqSw,
                                                     const float* __restrict__ wq_b,
                                                     float* __restrict__ n_q, int nNodes) {
  __shared__ __attribute__((aligned(16))) __bf16 ht[TILE][LDIM];
  int tid = threadIdx.x, lane = tid & 31, wave = tid >> 5;
  int n0 = blockIdx.x * TILE;
  bool full = (n0 + TILE) <= nNodes;
#pragma unroll
  for (int idx = tid; idx < TILE * LDIM / 4; idx += 128) {
    int m = idx >> 5, i4 = idx & 31;
    int n = n0 + m; if (n >= nNodes) n = nNodes - 1;
    float4 hv = ((const float4*)(h + (size_t)n * LDIM))[i4];
    v4bf bv = {(__bf16)hv.x, (__bf16)hv.y, (__bf16)hv.z, (__bf16)hv.w};
    *(v4bf*)&ht[m][i4 * 4] = bv;
  }
  __syncthreads();
  int laneHi = (lane >= 16) ? 1 : 0;
  int Mrow = lane & 15, N = lane & 15;
  for (int ct = 0; ct < 2; ++ct) {
    int col = (wave * 2 + ct) * 16;
    int ctIdx = col >> 4;
    v8f acc = {};
#pragma unroll
    for (int kb = 0; kb < 4; ++kb) {   // K = 128
      v16bf a = load_a_frag(&ht[0][0], &ht[0][0], Mrow, kb, laneHi);
      v16bf b = load_b_frag_sw(wqSw, kb * 8 + ctIdx, lane);
      acc = __builtin_amdgcn_wmma_f32_16x16x32_bf16(false, a, false, b,
                                                    (short)0, acc, false, false);
    }
    float bias = wq_b[col + N];
    if (full) {
#pragma unroll
      for (int r = 0; r < 8; ++r) {
        int M = r + 8 * laneHi;
        n_q[(size_t)(n0 + M) * LDIM + col + N] = acc[r] + bias;
      }
    } else {
#pragma unroll
      for (int r = 0; r < 8; ++r) {
        int M = r + 8 * laneHi;
        int n = n0 + M;
        if (n < nNodes) n_q[(size_t)n * LDIM + col + N] = acc[r] + bias;
      }
    }
  }
}

// qb[n] = wk_b . n_q[n];  self_y[n] = h[n].v_self + wv_b; zero accumulators
__global__ __launch_bounds__(256) void node_aux(const float* __restrict__ h,
                                                const float* __restrict__ n_q,
                                                const float* __restrict__ wk_b,
                                                const float* __restrict__ v_self,
                                                const float* __restrict__ wv_b,
                                                float* __restrict__ qb,
                                                float* __restrict__ self_y,
                                                float* wV0, float* z0,
                                                float* wV1, float* z1, int nNodes) {
  int n = blockIdx.x * 256 + threadIdx.x;
  if (n >= nNodes) return;
  float a = 0.f, s = 0.f;
  const float4* q4 = (const float4*)(n_q + (size_t)n * LDIM);
  const float4* h4 = (const float4*)(h + (size_t)n * LDIM);
  const float4* kb4 = (const float4*)wk_b;
  const float4* vs4 = (const float4*)v_self;
  for (int i = 0; i < LDIM / 4; ++i) {
    float4 qv = q4[i], kv = kb4[i], hv = h4[i], sv = vs4[i];
    a += kv.x * qv.x + kv.y * qv.y + kv.z * qv.z + kv.w * qv.w;
    s += hv.x * sv.x + hv.y * sv.y + hv.z * sv.z + hv.w * sv.w;
  }
  qb[n] = a;
  self_y[n] = s + wv_b[0];
  wV0[n] = 0.f; z0[n] = 0.f; wV1[n] = 0.f; z1[n] = 0.f;
}

// ---------------- Edge attention (hot kernel, WMMA) ----------------

__global__ __launch_bounds__(128) void edge_attn_kernel(
    const float* __restrict__ h, const int* __restrict__ src,
    const int* __restrict__ dst, const int* __restrict__ etype,
    const float* __restrict__ coef, const __bf16* __restrict__ WkSw,
    const float* __restrict__ Wv, const float* __restrict__ n_q,
    const float* __restrict__ qb, const float* __restrict__ wv_b,
    float* wV0, float* z0, float* wV1, float* z1, int nEdges) {
  __shared__ __attribute__((aligned(16))) __bf16 hs[TILE][LDIM];
  __shared__ __attribute__((aligned(16))) __bf16 hd[TILE][LDIM];
  __shared__ __attribute__((aligned(16))) float qs[TILE][LDIM];
  __shared__ __attribute__((aligned(16))) float qd[TILE][LDIM];
  __shared__ float  c0s[TILE], c1s[TILE];
  __shared__ int    sIdx[TILE], dIdx[TILE], validE[TILE];
  __shared__ float  scoreF[TILE], scoreR[TILE], nvF[TILE], nvR[TILE];

  int tid = threadIdx.x, lane = tid & 31, wave = tid >> 5;
  int e0 = blockIdx.x * TILE;

  if (tid < TILE) {
    int e = e0 + tid;
    int valid = (e < nEdges);
    if (!valid) e = nEdges - 1;
    int t = etype[e];
    sIdx[tid] = src[e]; dIdx[tid] = dst[e]; validE[tid] = valid;
    c0s[tid] = coef[t * 2 + 0]; c1s[tid] = coef[t * 2 + 1];
    scoreF[tid] = 0.f; scoreR[tid] = 0.f;
  }
  __syncthreads();
#pragma unroll
  for (int idx = tid; idx < TILE * LDIM / 4; idx += 128) {
    int m = idx >> 5, i4 = idx & 31;
    int s = sIdx[m], d = dIdx[m];
    float4 hvs = ((const float4*)(h + (size_t)s * LDIM))[i4];
    float4 hvd = ((const float4*)(h + (size_t)d * LDIM))[i4];
    v4bf bs = {(__bf16)hvs.x, (__bf16)hvs.y, (__bf16)hvs.z, (__bf16)hvs.w};
    v4bf bd = {(__bf16)hvd.x, (__bf16)hvd.y, (__bf16)hvd.z, (__bf16)hvd.w};
    *(v4bf*)&hs[m][i4 * 4] = bs;
    *(v4bf*)&hd[m][i4 * 4] = bd;
    ((float4*)&qs[m][0])[i4] = ((const float4*)(n_q + (size_t)s * LDIM))[i4];
    ((float4*)&qd[m][0])[i4] = ((const float4*)(n_q + (size_t)d * LDIM))[i4];
  }
  __syncthreads();

  // n_v: 8 threads per edge, 32 K each (4 chunks of 8), combined-basis Wv
  {
    int m = tid >> 3, part = tid & 7;
    float c0 = c0s[m], c1 = c1s[m];
    float accF = 0.f, accR = 0.f;
#pragma unroll
    for (int c = 0; c < 4; ++c) {
      int K = part * 32 + c * 8;
      const __bf16* pF = (K < LDIM) ? &hs[m][K] : &hd[m][K - LDIM];
      const __bf16* pR = (K < LDIM) ? &hd[m][K] : &hs[m][K - LDIM];
      v8bf zf = *(const v8bf*)pF;
      v8bf zr = *(const v8bf*)pR;
#pragma unroll
      for (int j2 = 0; j2 < 2; ++j2) {
        float4 wv0 = *(const float4*)(Wv + K + j2 * 4);
        float4 wv1 = *(const float4*)(Wv + 256 + K + j2 * 4);
        float w0 = c0 * wv0.x + c1 * wv1.x;
        float w1 = c0 * wv0.y + c1 * wv1.y;
        float w2 = c0 * wv0.z + c1 * wv1.z;
        float w3 = c0 * wv0.w + c1 * wv1.w;
        accF += (float)zf[j2 * 4 + 0] * w0 + (float)zf[j2 * 4 + 1] * w1 +
                (float)zf[j2 * 4 + 2] * w2 + (float)zf[j2 * 4 + 3] * w3;
        accR += (float)zr[j2 * 4 + 0] * w0 + (float)zr[j2 * 4 + 1] * w1 +
                (float)zr[j2 * 4 + 2] * w2 + (float)zr[j2 * 4 + 3] * w3;
      }
    }
    for (int off = 4; off; off >>= 1) {
      accF += __shfl_down(accF, off, 32);
      accR += __shfl_down(accR, off, 32);
    }
    if (part == 0) { nvF[m] = accF; nvR[m] = accR; }
  }

  // n_k via WMMA: B fragments are direction-independent -> share them
  int laneHi = (lane >= 16) ? 1 : 0;
  int Mrow = lane & 15, N = lane & 15;
  for (int ct = 0; ct < 2; ++ct) {
    int col = (wave * 2 + ct) * 16;
    int ctIdx = col >> 4;
    v8f a0F = {}, a1F = {}, a0R = {}, a1R = {};
#pragma unroll
    for (int kb = 0; kb < 8; ++kb) {          // K = 256
      v16bf b0 = load_b_frag_sw(WkSw, (0 * 8 + kb) * 8 + ctIdx, lane);
      v16bf b1 = load_b_frag_sw(WkSw, (1 * 8 + kb) * 8 + ctIdx, lane);
      v16bf aF = load_a_frag(&hs[0][0], &hd[0][0], Mrow, kb, laneHi);
      v16bf aR = load_a_frag(&hd[0][0], &hs[0][0], Mrow, kb, laneHi);
      a0F = __builtin_amdgcn_wmma_f32_16x16x32_bf16(false, aF, false, b0,
                                                    (short)0, a0F, false, false);
      a1F = __builtin_amdgcn_wmma_f32_16x16x32_bf16(false, aF, false, b1,
                                                    (short)0, a1F, false, false);
      a0R = __builtin_amdgcn_wmma_f32_16x16x32_bf16(false, aR, false, b0,
                                                    (short)0, a0R, false, false);
      a1R = __builtin_amdgcn_wmma_f32_16x16x32_bf16(false, aR, false, b1,
                                                    (short)0, a1R, false, false);
    }
#pragma unroll
    for (int r = 0; r < 8; ++r) {
      int M = r + 8 * laneHi;
      float c0 = c0s[M], c1 = c1s[M];
      float pF = (c0 * a0F[r] + c1 * a1F[r]) * qd[M][col + N];
      float pR = (c0 * a0R[r] + c1 * a1R[r]) * qs[M][col + N];
      pF += __shfl_xor(pF, 1, 32); pR += __shfl_xor(pR, 1, 32);
      pF += __shfl_xor(pF, 2, 32); pR += __shfl_xor(pR, 2, 32);
      pF += __shfl_xor(pF, 4, 32); pR += __shfl_xor(pR, 4, 32);
      pF += __shfl_xor(pF, 8, 32); pR += __shfl_xor(pR, 8, 32);
      if (N == 0) {
        atomicAdd(&scoreF[M], pF);
        atomicAdd(&scoreR[M], pR);
      }
    }
  }
  __syncthreads();

  if (tid < TILE && validE[tid]) {
    const float invScale = 0.08838834764831845f;   // 1/sqrt(128)
    float vb = wv_b[0];
    int m = tid;
    {
      int t = dIdx[m];
      float s = (scoreF[m] + qb[t]) * invScale;
      s = __expf(fminf(fmaxf(s, -10.f), 10.f));
      atomicAdd(&wV0[t], s * (nvF[m] + vb));
      atomicAdd(&z0[t], s);
    }
    {
      int t = sIdx[m];
      float s = (scoreR[m] + qb[t]) * invScale;
      s = __expf(fminf(fmaxf(s, -10.f), 10.f));
      atomicAdd(&wV1[t], s * (nvR[m] + vb));
      atomicAdd(&z1[t], s);
    }
  }
}

// ---------------- y, keys, top-k ----------------

__global__ __launch_bounds__(256) void finalize_y(const float* wV0, const float* z0,
                                                  const float* wV1, const float* z1,
                                                  const float* self_y, float* y,
                                                  unsigned* keys, int nNodes) {
  int n = blockIdx.x * 256 + threadIdx.x;
  if (n >= nNodes) return;
  float v = wV0[n] / (z0[n] + 1e-6f) + wV1[n] / (z1[n] + 1e-6f) + self_y[n];
  y[n] = v;
  unsigned u = __float_as_uint(v);
  keys[n] = (u & 0x80000000u) ? ~u : (u | 0x80000000u);   // ascending-orderable
}

__global__ void topk_init(unsigned* hist, int* st, int K) {
  int i = threadIdx.x;
  if (i < 256) hist[i] = 0u;
  if (i == 0) { st[0] = 0; st[1] = K; }
}

__global__ __launch_bounds__(256) void topk_hist(const unsigned* keys, const int* st,
                                                 unsigned* hist, int pass, int n) {
  int i = blockIdx.x * 256 + threadIdx.x;
  if (i >= n) return;
  unsigned k = keys[i];
  if (pass > 0 && (k >> (32 - 8 * pass)) != (unsigned)st[0]) return;
  atomicAdd(&hist[(k >> (24 - 8 * pass)) & 255u], 1u);
}

__global__ void topk_select(unsigned* hist, int* st) {
  if (threadIdx.x == 0) {
    int rem = st[1];
    unsigned acc = 0;
    int chosen = 0;
    for (int d = 255; d >= 0; --d) {
      unsigned c = hist[d];
      if (acc + c >= (unsigned)rem) { chosen = d; st[1] = rem - (int)acc; break; }
      acc += c;
    }
    st[0] = (st[0] << 8) | chosen;
  }
  __syncthreads();
  for (int i = threadIdx.x; i < 256; i += blockDim.x) hist[i] = 0u;
}

__global__ __launch_bounds__(256) void topk_flags(const unsigned* keys, const int* st,
                                                  int* gt, int* eq, int n) {
  int i = blockIdx.x * 256 + threadIdx.x;
  if (i >= n) return;
  unsigned thr = (unsigned)st[0], k = keys[i];
  gt[i] = (k > thr) ? 1 : 0;
  eq[i] = (k == thr) ? 1 : 0;
}

__global__ __launch_bounds__(256) void scan_block_reduce(const int* flags, int* blockSums, int n) {
  __shared__ int sm[256];
  int i = blockIdx.x * 256 + threadIdx.x;
  sm[threadIdx.x] = (i < n) ? flags[i] : 0;
  __syncthreads();
  for (int off = 128; off; off >>= 1) {
    if (threadIdx.x < off) sm[threadIdx.x] += sm[threadIdx.x + off];
    __syncthreads();
  }
  if (threadIdx.x == 0) blockSums[blockIdx.x] = sm[0];
}

__global__ void scan_exclusive_small(int* blockSums, int nb) {
  if (threadIdx.x == 0 && blockIdx.x == 0) {
    int run = 0;
    for (int b = 0; b < nb; ++b) { int v = blockSums[b]; blockSums[b] = run; run += v; }
  }
}

__global__ __launch_bounds__(256) void scan_apply(const int* flags, const int* blockOffs,
                                                  int* prefix, int n) {
  __shared__ int sm[256];
  int i = blockIdx.x * 256 + threadIdx.x;
  int v = (i < n) ? flags[i] : 0;
  sm[threadIdx.x] = v;
  __syncthreads();
  for (int off = 1; off < 256; off <<= 1) {
    int t = (threadIdx.x >= off) ? sm[threadIdx.x - off] : 0;
    __syncthreads();
    sm[threadIdx.x] += t;
    __syncthreads();
  }
  if (i < n) prefix[i] = blockOffs[blockIdx.x] + sm[threadIdx.x] - v;  // exclusive
}

__global__ __launch_bounds__(256) void make_sel(const int* gt, const int* eq,
                                                const int* eqPre, const int* st,
                                                int* sel, int n) {
  int i = blockIdx.x * 256 + threadIdx.x;
  if (i >= n) return;
  sel[i] = (gt[i] || (eq[i] && eqPre[i] < st[1])) ? 1 : 0;
}

__global__ __launch_bounds__(256) void scatter_topk(const int* sel, const int* pos,
                                                    const float* h, const float* y,
                                                    int* idsOut, float* hOut, int n) {
  int i = blockIdx.x * 256 + threadIdx.x;
  if (i >= n || !sel[i]) return;
  int k = pos[i];
  idsOut[k] = i;
  float sg = 1.f / (1.f + __expf(-y[i]));
  const float4* hp = (const float4*)(h + (size_t)i * LDIM);
  float4* op = (float4*)(hOut + (size_t)k * LDIM);
  for (int j = 0; j < LDIM / 4; ++j) {
    float4 v = hp[j];
    v.x *= sg; v.y *= sg; v.z *= sg; v.w *= sg;
    op[j] = v;
  }
}

// ---------------- Host orchestration ----------------

extern "C" void kernel_launch(void* const* d_in, const int* in_sizes, int n_in,
                              void* d_out, int out_size, void* d_ws, size_t ws_size,
                              hipStream_t stream) {
  const float* h     = (const float*)d_in[0];
  const int*   src   = (const int*)d_in[1];
  const int*   dst   = (const int*)d_in[2];
  const int*   etype = (const int*)d_in[3];
  const float* Wb    = (const float*)d_in[4];
  const float* coef  = (const float*)d_in[5];
  const float* wq_w  = (const float*)d_in[6];
  const float* wq_b  = (const float*)d_in[7];
  const float* wk_w  = (const float*)d_in[8];
  const float* wk_b  = (const float*)d_in[9];
  const float* wv_w  = (const float*)d_in[10];
  const float* wv_b  = (const float*)d_in[11];

  int nNodes = in_sizes[0] / LDIM;
  int nEdges = in_sizes[1];
  int K = nNodes / 2; if (K < 2) K = 2;

  char* p = (char*)d_ws;
  auto alloc = [&](size_t bytes) -> void* {
    void* r = (void*)p;
    p += (bytes + 255) & ~((size_t)255);
    return r;
  };
  __bf16* WkSw   = (__bf16*)alloc((size_t)2 * 256 * LDIM * sizeof(__bf16));
  __bf16* wqSw   = (__bf16*)alloc((size_t)LDIM * LDIM * sizeof(__bf16));
  float*  Wv     = (float*)alloc(2 * 256 * sizeof(float));
  float*  v_self = (float*)alloc(LDIM * sizeof(float));
  float*  n_q    = (float*)alloc((size_t)nNodes * LDIM * sizeof(float));
  float*  qb     = (float*)alloc((size_t)nNodes * sizeof(float));
  float*  self_y = (float*)alloc((size_t)nNodes * sizeof(float));
  float*  wV0    = (float*)alloc((size_t)nNodes * sizeof(float));
  float*  z0     = (float*)alloc((size_t)nNodes * sizeof(float));
  float*  wV1    = (float*)alloc((size_t)nNodes * sizeof(float));
  float*  z1     = (float*)alloc((size_t)nNodes * sizeof(float));
  float*  y      = (float*)alloc((size_t)nNodes * sizeof(float));
  unsigned* keys = (unsigned*)alloc((size_t)nNodes * sizeof(unsigned));
  unsigned* hist = (unsigned*)alloc(256 * sizeof(unsigned));
  int*    st     = (int*)alloc(8 * sizeof(int));
  int*    gt     = (int*)alloc((size_t)nNodes * sizeof(int));
  int*    eq     = (int*)alloc((size_t)nNodes * sizeof(int));
  int*    eqPre  = (int*)alloc((size_t)nNodes * sizeof(int));
  int*    sel    = (int*)alloc((size_t)nNodes * sizeof(int));
  int*    selPre = (int*)alloc((size_t)nNodes * sizeof(int));
  int*    bSums  = (int*)alloc(1024 * sizeof(int));

  int nB256  = (nNodes + 255) / 256;
  int nTiles = (nNodes + TILE - 1) / TILE;
  int eTiles = (nEdges + TILE - 1) / TILE;

  build_wk<<<512, 128, 0, stream>>>(Wb, wk_w, wv_w, WkSw, Wv);
  prep_misc<<<(LDIM * LDIM + 255) / 256, 256, 0, stream>>>(wq_w, wqSw, Wv, coef, v_self);
  node_q_kernel<<<nTiles, 128, 0, stream>>>(h, wqSw, wq_b, n_q, nNodes);
  node_aux<<<nB256, 256, 0, stream>>>(h, n_q, wk_b, v_self, wv_b, qb, self_y,
                                      wV0, z0, wV1, z1, nNodes);
  edge_attn_kernel<<<eTiles, 128, 0, stream>>>(h, src, dst, etype, coef, WkSw, Wv,
                                               n_q, qb, wv_b, wV0, z0, wV1, z1, nEdges);
  finalize_y<<<nB256, 256, 0, stream>>>(wV0, z0, wV1, z1, self_y, y, keys, nNodes);

  topk_init<<<1, 256, 0, stream>>>(hist, st, K);
  for (int pass = 0; pass < 4; ++pass) {
    topk_hist<<<nB256, 256, 0, stream>>>(keys, st, hist, pass, nNodes);
    topk_select<<<1, 256, 0, stream>>>(hist, st);
  }
  topk_flags<<<nB256, 256, 0, stream>>>(keys, st, gt, eq, nNodes);
  scan_block_reduce<<<nB256, 256, 0, stream>>>(eq, bSums, nNodes);
  scan_exclusive_small<<<1, 1, 0, stream>>>(bSums, nB256);
  scan_apply<<<nB256, 256, 0, stream>>>(eq, bSums, eqPre, nNodes);
  make_sel<<<nB256, 256, 0, stream>>>(gt, eq, eqPre, st, sel, nNodes);
  scan_block_reduce<<<nB256, 256, 0, stream>>>(sel, bSums, nNodes);
  scan_exclusive_small<<<1, 1, 0, stream>>>(bSums, nB256);
  scan_apply<<<nB256, 256, 0, stream>>>(sel, bSums, selPre, nNodes);

  int*   idsOut = (int*)d_out;                 // first K elements: node ids
  float* hOut   = (float*)d_out + K;           // then K*128 floats: updated_h
  scatter_topk<<<nB256, 256, 0, stream>>>(sel, selPre, h, y, idsOut, hOut, nNodes);
}
